// Model_8022998909301
// MI455X (gfx1250) — matrix operand order repaired and checked by reading, not compiled, not run on hardware
//
#include <hip/hip_runtime.h>
#include <hip/hip_bf16.h>

typedef __attribute__((ext_vector_type(16))) _Float16 v16h;
typedef __attribute__((ext_vector_type(8)))  _Float16 v8h;
typedef __attribute__((ext_vector_type(4)))  _Float16 v4h;
typedef __attribute__((ext_vector_type(8)))  float    v8f;
typedef int v4i_ __attribute__((vector_size(16)));

#define AS1 __attribute__((address_space(1)))
#define AS3 __attribute__((address_space(3)))

#if defined(__gfx1250__) && __has_builtin(__builtin_amdgcn_global_load_async_to_lds_b128) && __has_builtin(__builtin_amdgcn_s_wait_asynccnt)
#define USE_ASYNC_LDS 1
#else
#define USE_ASYNC_LDS 0
#endif

// ---------------------------------------------------------------------------
// A-fetch functors: load8(m, k8) returns 8 consecutive-K f16 values for row m,
// k8 a multiple of 8 (matches WMMA 16-bit A layout: runs at k0+half*8, +16).
// ---------------------------------------------------------------------------
struct Conv1A {                 // f16 image 3x84x84, 8x8 s4 -> 20x20; K=ci*64+ky*8+kx
  const _Float16* x;
  __device__ __forceinline__ v8h load8(int m, int k8) const {
    int b  = m / 400; int r = m - b * 400;
    int oy = r / 20;  int ox = r - oy * 20;
    int ci = k8 >> 6; int ky = (k8 >> 3) & 7;            // kx = 0 for the run
    const _Float16* q = x + (((b * 3 + ci) * 84) + oy * 4 + ky) * 84 + ox * 4;
    v4h lo = *(const v4h*)q;                             // 8B-aligned b64
    v4h hi = *(const v4h*)(q + 4);
    v8h v;
#pragma unroll
    for (int j = 0; j < 4; ++j) { v[j] = lo[j]; v[j + 4] = hi[j]; }
    return v;
  }
};

struct Conv2A {                 // 32x20x20, 4x4 s2 -> 9x9; K = ci*16+ky*4+kx
  const _Float16* x;
  __device__ __forceinline__ v8h load8(int m, int k8) const {
    int b  = m / 81; int r = m - b * 81;
    int oy = r / 9;  int ox = r - oy * 9;
    int ci = k8 >> 4; int ky = (k8 >> 2) & 3;            // run = rows ky,ky+1
    const _Float16* q = x + ((b * 32 + ci) * 20 + oy * 2 + ky) * 20 + ox * 2;
    v8h v;
#pragma unroll
    for (int j = 0; j < 4; ++j) { v[j] = q[j]; v[j + 4] = q[20 + j]; }
    return v;
  }
};

struct Conv3A {                 // 64x9x9, 3x3 s1 -> 7x7; channel-minor K = s*64+ci
  const _Float16* x;
  __device__ __forceinline__ v8h load8(int m, int k8) const {
    int b  = m / 49; int r = m - b * 49;
    int oy = r / 7;  int ox = r - oy * 7;
    int ci = k8 & 63; int s = k8 >> 6;                   // run stays in one s
    int ky = s / 3;   int kx = s - ky * 3;
    const _Float16* q = x + ((b * 64 + ci) * 9 + oy + ky) * 9 + ox + kx;
    v8h v;
#pragma unroll
    for (int j = 0; j < 8; ++j) v[j] = q[j * 81];        // stride = channel plane
    return v;
  }
};

struct DenseA {
  const _Float16* x; int K;
  __device__ __forceinline__ v8h load8(int m, int k8) const {
    return *(const v8h*)(x + (size_t)m * K + k8);        // 16B-aligned b128
  }
};

struct Mlp1A {                  // concat([mu, ls, one_hot(a)], pad 98->128)
  const float* z; const int* act;
  __device__ __forceinline__ v8h load8(int m, int k8) const {
    v8h v;
    if (k8 < 80) {
      const float* q = z + m * 80 + k8;
#pragma unroll
      for (int j = 0; j < 8; ++j) v[j] = (_Float16)q[j];
    } else {
      int a = act[m];
#pragma unroll
      for (int j = 0; j < 8; ++j) {
        int kk = k8 + j;
        v[j] = (_Float16)((kk < 98 && a == kk - 80) ? 1.0f : 0.0f);
      }
    }
    return v;
  }
};

// ---------------------------------------------------------------------------
// Store functors
// ---------------------------------------------------------------------------
template <int COUT, int HOUT, int WOUT>
struct ConvSt {
  _Float16* y;
  __device__ __forceinline__ void operator()(int m, int n, float v) const {
    constexpr int HW = HOUT * WOUT;
    int b = m / HW; int r = m - b * HW;
    y[((b * COUT + n) * HOUT + r / WOUT) * WOUT + (r % WOUT)] = (_Float16)v;
  }
};
struct StH { _Float16* y; int N;
  __device__ __forceinline__ void operator()(int m, int n, float v) const { y[m * N + n] = (_Float16)v; } };
struct StF { float* y; int N;
  __device__ __forceinline__ void operator()(int m, int n, float v) const { y[m * N + n] = v; } };

// ---------------------------------------------------------------------------
// WMMA GEMM engine.  B pre-transposed [N][K]; chunk async-staged into LDS as
// Bs[N][KCHUNK].  Each wave computes WN 16x16 tiles along N, sharing one A
// fragment across WN independent accumulator chains (fills WMMA hazard slots).
// ---------------------------------------------------------------------------
template <int KDIM, int NDIM, int KCHUNK, int WN, bool RELU, typename AF, typename ST>
__global__ __launch_bounds__(256)
void gemm_wmma(AF af, const _Float16* __restrict__ Bw, const float* __restrict__ bias,
               ST st, int Mtiles) {
  static_assert(KCHUNK % 32 == 0 && KDIM % KCHUNK == 0, "k shape");
  static_assert(NDIM % (16 * WN) == 0, "n shape");
  __shared__ _Float16 Bs[NDIM * KCHUNK];
  constexpr int NTP  = NDIM / (16 * WN);   // N tile-groups
  constexpr int KV   = KCHUNK / 8;
  constexpr int VECS = NDIM * KV;

  const int lane = threadIdx.x & 31;
  const int wave = threadIdx.x >> 5;
  int tile = blockIdx.x * 8 + wave;
  int mt  = tile / NTP;
  int ntp = tile - mt * NTP;
  const bool active = (mt < Mtiles);
  if (!active) { mt = 0; ntp = 0; }

  const int half = lane >> 4;
  const int l15  = lane & 15;
  const int arow = mt * 16 + l15;

  v8f acc[WN];
#pragma unroll
  for (int w = 0; w < WN; ++w) acc[w] = (v8f){};

  for (int kc = 0; kc < KDIM; kc += KCHUNK) {
    __syncthreads();
#if USE_ASYNC_LDS
    for (int v = threadIdx.x; v < VECS; v += 256) {
      const int n  = v / KV;
      const int ko = (v - n * KV) * 8;
      const _Float16* src = Bw + (size_t)n * KDIM + kc + ko;
      __builtin_amdgcn_global_load_async_to_lds_b128(
          (AS1 v4i_*)(void*)src, (AS3 v4i_*)(void*)&Bs[n * KCHUNK + ko], 0, 0);
    }
    __builtin_amdgcn_s_wait_asynccnt(0);
#else
    for (int v = threadIdx.x; v < VECS; v += 256) {
      const int n  = v / KV;
      const int ko = (v - n * KV) * 8;
      *(v8h*)&Bs[n * KCHUNK + ko] = *(const v8h*)(Bw + (size_t)n * KDIM + kc + ko);
    }
#endif
    __syncthreads();

#pragma unroll
    for (int ks = 0; ks < KCHUNK; ks += 32) {
      const int ka = kc + ks + half * 8;
      v8h alo = af.load8(arow, ka);
      v8h ahi = af.load8(arow, ka + 16);
      v16h a;
#pragma unroll
      for (int j = 0; j < 8; ++j) { a[j] = alo[j]; a[j + 8] = ahi[j]; }
#pragma unroll
      for (int w = 0; w < WN; ++w) {
        const int bco = (ntp * WN + w) * 16 + l15;
        const _Float16* bp = &Bs[bco * KCHUNK + ks + half * 8];
        v8h blo = *(const v8h*)bp;          // ds_load_b128
        v8h bhi = *(const v8h*)(bp + 16);   // ds_load_b128
        v16h b;
#pragma unroll
        for (int j = 0; j < 8; ++j) { b[j] = blo[j]; b[j + 8] = bhi[j]; }
        acc[w] = __builtin_amdgcn_wmma_f32_16x16x32_f16(false, a, false, b,
                                                        (short)0, acc[w], false, false);
      }
    }
  }

  if (active) {
#pragma unroll
    for (int w = 0; w < WN; ++w) {
      const int n = (ntp * WN + w) * 16 + l15;
      const float bb = bias[n];
#pragma unroll
      for (int r = 0; r < 8; ++r) {
        const int m = mt * 16 + r + (half << 3);
        float v = acc[w][r] + bb;
        if (RELU) v = fmaxf(v, 0.0f);
        st(m, n, v);
      }
    }
  }
}

// ---------------------------------------------------------------------------
// Prep kernels
// ---------------------------------------------------------------------------
__global__ void x_to_f16(const float* __restrict__ x, _Float16* __restrict__ y, int n4) {
  int i = blockIdx.x * 256 + threadIdx.x;
  if (i < n4) {
    float4 v = ((const float4*)x)[i];
    v4h o; o[0] = (_Float16)v.x; o[1] = (_Float16)v.y;
    o[2] = (_Float16)v.z; o[3] = (_Float16)v.w;
    *(v4h*)(y + (size_t)i * 4) = o;
  }
}

__global__ void w_to_f16(const float* __restrict__ w, _Float16* __restrict__ B, int n) {
  int i = blockIdx.x * 256 + threadIdx.x;
  if (i < n) B[i] = (_Float16)w[i];                 // OIHW is already [N][K] ch-major
}

__global__ void conv3_w_cmin(const float* __restrict__ w, _Float16* __restrict__ B) {
  int i = blockIdx.x * 256 + threadIdx.x;           // B[n*576 + s*64 + ci]
  if (i < 64 * 576) {
    int n = i / 576; int k = i - n * 576;
    int ci = k & 63; int s = k >> 6;
    B[i] = (_Float16)w[(n * 64 + ci) * 9 + s];
  }
}

__global__ void kn_to_nk(const float* __restrict__ w, _Float16* __restrict__ B,
                         int K, int N, int Kpad) {
  int i = blockIdx.x * 256 + threadIdx.x;
  if (i < N * Kpad) {
    int n = i / Kpad; int k = i - n * Kpad;
    B[i] = (k < K) ? (_Float16)w[k * N + n] : (_Float16)0.0f;
  }
}

__global__ void head_merge(const float* __restrict__ mw, const float* __restrict__ lw,
                           const float* __restrict__ mb, const float* __restrict__ lb,
                           _Float16* __restrict__ B, float* __restrict__ bias) {
  int i = blockIdx.x * 256 + threadIdx.x;
  if (i < 80 * 256) {
    int n = i / 256; int k = i - n * 256;
    B[i] = (_Float16)((n < 40) ? mw[k * 40 + n] : lw[k * 40 + (n - 40)]);
  }
  if (i < 80) bias[i] = (i < 40) ? mb[i] : lb[i - 40];
}

// ---------------------------------------------------------------------------
// KL(target || predicted), mean over batch
// ---------------------------------------------------------------------------
__global__ void zero_out(float* out) { if (threadIdx.x == 0) out[0] = 0.0f; }

__global__ void kl_kernel(const float* __restrict__ ztgt, const float* __restrict__ zp,
                          float* __restrict__ out) {
  __shared__ float red[256];
  float s = 0.0f;
  for (int i = blockIdx.x * 256 + threadIdx.x; i < 1024 * 40; i += gridDim.x * 256) {
    int b = i / 40, j = i - b * 40;
    float mu_t = ztgt[b * 80 + j];
    float ls_t = ztgt[b * 80 + 40 + j];
    float mu_p = zp[b * 80 + j];
    float ls_p = zp[b * 80 + 40 + j];
    float var_p = expf(ls_t) + 1e-8f;
    float var_q = expf(ls_p) + 1e-8f;
    float d = mu_p - mu_t;
    s += 0.5f * (var_p / var_q + d * d / var_q - 1.0f + logf(var_q) - logf(var_p));
  }
  red[threadIdx.x] = s;
  __syncthreads();
  for (int o = 128; o > 0; o >>= 1) {
    if (threadIdx.x < o) red[threadIdx.x] += red[threadIdx.x + o];
    __syncthreads();
  }
  if (threadIdx.x == 0) atomicAdd(out, red[0] * (1.0f / 1024.0f));
}

// ---------------------------------------------------------------------------
// Launch
// ---------------------------------------------------------------------------
extern "C" void kernel_launch(void* const* d_in, const int* in_sizes, int n_in,
                              void* d_out, int out_size, void* d_ws, size_t ws_size,
                              hipStream_t stream) {
  const float* x_t    = (const float*)d_in[0];
  const float* x_tp1  = (const float*)d_in[1];
  const int*   a_t    = (const int*)d_in[2];

  const float* enc_w1 = (const float*)d_in[3];  const float* enc_b1 = (const float*)d_in[4];
  const float* enc_w2 = (const float*)d_in[5];  const float* enc_b2 = (const float*)d_in[6];
  const float* enc_w3 = (const float*)d_in[7];  const float* enc_b3 = (const float*)d_in[8];
  const float* enc_fw = (const float*)d_in[9];  const float* enc_fb = (const float*)d_in[10];
  const float* tgt_w1 = (const float*)d_in[11]; const float* tgt_b1 = (const float*)d_in[12];
  const float* tgt_w2 = (const float*)d_in[13]; const float* tgt_b2 = (const float*)d_in[14];
  const float* tgt_w3 = (const float*)d_in[15]; const float* tgt_b3 = (const float*)d_in[16];
  const float* tgt_fw = (const float*)d_in[17]; const float* tgt_fb = (const float*)d_in[18];
  const float* wm_w1  = (const float*)d_in[19]; const float* wm_b1  = (const float*)d_in[20];
  const float* wm_w2  = (const float*)d_in[21]; const float* wm_b2  = (const float*)d_in[22];
  const float* wm_w3  = (const float*)d_in[23]; const float* wm_b3  = (const float*)d_in[24];
  const float* wm_mw  = (const float*)d_in[25]; const float* wm_mb  = (const float*)d_in[26];
  const float* wm_lw  = (const float*)d_in[27]; const float* wm_lb  = (const float*)d_in[28];

  char* p = (char*)d_ws;
  auto alloc = [&](size_t bytes) -> void* {
    void* r = (void*)p;
    p += (bytes + 255) & ~(size_t)255;
    return r;
  };

  _Float16* wB1e = (_Float16*)alloc(32 * 192 * 2);
  _Float16* wB2e = (_Float16*)alloc(64 * 512 * 2);
  _Float16* wB3e = (_Float16*)alloc(64 * 576 * 2);
  _Float16* wBfe = (_Float16*)alloc((size_t)80 * 3136 * 2);
  _Float16* wB1t = (_Float16*)alloc(32 * 192 * 2);
  _Float16* wB2t = (_Float16*)alloc(64 * 512 * 2);
  _Float16* wB3t = (_Float16*)alloc(64 * 576 * 2);
  _Float16* wBft = (_Float16*)alloc((size_t)80 * 3136 * 2);
  _Float16* wBm1 = (_Float16*)alloc(256 * 128 * 2);
  _Float16* wBm2 = (_Float16*)alloc(256 * 256 * 2);
  _Float16* wBm3 = (_Float16*)alloc(256 * 256 * 2);
  _Float16* wBh  = (_Float16*)alloc(80 * 256 * 2);
  float*    bh   = (float*)alloc(80 * 4);
  _Float16* xh   = (_Float16*)alloc((size_t)1024 * 3 * 84 * 84 * 2);
  _Float16* h1   = (_Float16*)alloc((size_t)1024 * 32 * 400 * 2);
  _Float16* h2   = (_Float16*)alloc((size_t)1024 * 64 * 81 * 2);
  _Float16* h3   = (_Float16*)alloc((size_t)1024 * 64 * 49 * 2);
  float*    z_e  = (float*)alloc((size_t)1024 * 80 * 4);
  float*    z_t  = (float*)alloc((size_t)1024 * 80 * 4);
  _Float16* m1   = (_Float16*)alloc((size_t)1024 * 256 * 2);
  _Float16* m2   = (_Float16*)alloc((size_t)1024 * 256 * 2);
  _Float16* m3   = (_Float16*)alloc((size_t)1024 * 256 * 2);
  float*    zp   = (float*)alloc((size_t)1024 * 80 * 4);

  auto blk = [](int n) { return dim3((n + 255) / 256); };

  // ---- weight prep ----
  w_to_f16<<<blk(32 * 192), 256, 0, stream>>>(enc_w1, wB1e, 32 * 192);
  w_to_f16<<<blk(64 * 512), 256, 0, stream>>>(enc_w2, wB2e, 64 * 512);
  conv3_w_cmin<<<blk(64 * 576), 256, 0, stream>>>(enc_w3, wB3e);
  kn_to_nk<<<blk(80 * 3136), 256, 0, stream>>>(enc_fw, wBfe, 3136, 80, 3136);
  w_to_f16<<<blk(32 * 192), 256, 0, stream>>>(tgt_w1, wB1t, 32 * 192);
  w_to_f16<<<blk(64 * 512), 256, 0, stream>>>(tgt_w2, wB2t, 64 * 512);
  conv3_w_cmin<<<blk(64 * 576), 256, 0, stream>>>(tgt_w3, wB3t);
  kn_to_nk<<<blk(80 * 3136), 256, 0, stream>>>(tgt_fw, wBft, 3136, 80, 3136);
  kn_to_nk<<<blk(256 * 128), 256, 0, stream>>>(wm_w1, wBm1, 98, 256, 128);
  kn_to_nk<<<blk(256 * 256), 256, 0, stream>>>(wm_w2, wBm2, 256, 256, 256);
  kn_to_nk<<<blk(256 * 256), 256, 0, stream>>>(wm_w3, wBm3, 256, 256, 256);
  head_merge<<<blk(80 * 256), 256, 0, stream>>>(wm_mw, wm_lw, wm_mb, wm_lb, wBh, bh);

  using S1 = ConvSt<32, 20, 20>;
  using S2 = ConvSt<64, 9, 9>;
  using S3 = ConvSt<64, 7, 7>;

  const int MT1 = 1024 * 400 / 16;  // 25600
  const int MT2 = 1024 * 81 / 16;   // 5184
  const int MT3 = 1024 * 49 / 16;   // 3136
  const int MTB = 1024 / 16;        // 64
  const int NX4 = 1024 * 3 * 84 * 84 / 4;
  auto grid = [](int tiles) { return dim3((tiles + 7) / 8); };

  // ---- encoder(x_t) ----
  x_to_f16<<<blk(NX4), 256, 0, stream>>>(x_t, xh, NX4);
  gemm_wmma<192, 32, 192, 2, true, Conv1A, S1><<<grid(MT1), 256, 0, stream>>>(Conv1A{xh}, wB1e, enc_b1, S1{h1}, MT1);
  gemm_wmma<512, 64, 128, 4, true, Conv2A, S2><<<grid(MT2), 256, 0, stream>>>(Conv2A{h1}, wB2e, enc_b2, S2{h2}, MT2);
  gemm_wmma<576, 64,  96, 4, true, Conv3A, S3><<<grid(MT3), 256, 0, stream>>>(Conv3A{h2}, wB3e, enc_b3, S3{h3}, MT3);
  gemm_wmma<3136, 80, 224, 1, false, DenseA, StF><<<grid(MTB * 5), 256, 0, stream>>>(DenseA{h3, 3136}, wBfe, enc_fb, StF{z_e, 80}, MTB);

  // ---- target encoder(x_tp1) ----
  x_to_f16<<<blk(NX4), 256, 0, stream>>>(x_tp1, xh, NX4);
  gemm_wmma<192, 32, 192, 2, true, Conv1A, S1><<<grid(MT1), 256, 0, stream>>>(Conv1A{xh}, wB1t, tgt_b1, S1{h1}, MT1);
  gemm_wmma<512, 64, 128, 4, true, Conv2A, S2><<<grid(MT2), 256, 0, stream>>>(Conv2A{h1}, wB2t, tgt_b2, S2{h2}, MT2);
  gemm_wmma<576, 64,  96, 4, true, Conv3A, S3><<<grid(MT3), 256, 0, stream>>>(Conv3A{h2}, wB3t, tgt_b3, S3{h3}, MT3);
  gemm_wmma<3136, 80, 224, 1, false, DenseA, StF><<<grid(MTB * 5), 256, 0, stream>>>(DenseA{h3, 3136}, wBft, tgt_fb, StF{z_t, 80}, MTB);

  // ---- world model MLP ----
  gemm_wmma<128, 256, 64, 4, true, Mlp1A,  StH><<<grid(MTB * 4), 256, 0, stream>>>(Mlp1A{z_e, a_t}, wBm1, wm_b1, StH{m1, 256}, MTB);
  gemm_wmma<256, 256, 64, 4, true, DenseA, StH><<<grid(MTB * 4), 256, 0, stream>>>(DenseA{m1, 256}, wBm2, wm_b2, StH{m2, 256}, MTB);
  gemm_wmma<256, 256, 64, 4, true, DenseA, StH><<<grid(MTB * 4), 256, 0, stream>>>(DenseA{m2, 256}, wBm3, wm_b3, StH{m3, 256}, MTB);
  gemm_wmma<256, 80, 128, 1, false, DenseA, StF><<<grid(MTB * 5), 256, 0, stream>>>(DenseA{m3, 256}, wBh, bh, StF{zp, 80}, MTB);

  // ---- KL reduction ----
  zero_out<<<1, 32, 0, stream>>>((float*)d_out);
  kl_kernel<<<64, 256, 0, stream>>>(z_t, zp, (float*)d_out);
}
